// GCNEncoder_14929306321512
// MI455X (gfx1250) — compile-verified, matrix operand order checked by reading, not run on hardware
//
#include <hip/hip_runtime.h>
#include <hip/hip_bf16.h>

// ---------------- problem constants ----------------
#define N_NODES   50000
#define C         64
#define C2        128
#define E_EDGES   800000
#define N_AGGR    2
#define EPS_MSG   1e-7f
#define BN_EPS    1e-5f
#define DEN_EPS   1e-16f

typedef __attribute__((ext_vector_type(16))) __bf16 v16bf;
typedef __attribute__((ext_vector_type(8)))  float  v8f;

// f32 -> bf16, round-to-nearest-even, no reliance on float->__bf16 cast support
__device__ __forceinline__ __bf16 f2bf(float f) {
    union { float f; unsigned int u; } x; x.f = f;
    unsigned int r = (x.u + 0x7FFFu + ((x.u >> 16) & 1u)) >> 16;
    unsigned short hs = (unsigned short)r;
    __bf16 o;
    __builtin_memcpy(&o, &hs, 2);
    return o;
}

// ISA 16-bit WMMA A/B fragment K mapping (16x32 / 32x16 tiles, wave32)
__device__ __forceinline__ int fragK(int e, int half) {
    int kq = e >> 1;
    return (kq < 4 ? 2 * kq : 16 + 2 * (kq - 4)) + (half ? 8 : 0) + (e & 1);
}

// float atomic max via int-max / uint-min trick (valid with -inf init)
__device__ __forceinline__ void atomic_max_f32(float* addr, float v) {
    if (v >= 0.0f) atomicMax((int*)addr, __float_as_int(v));
    else           atomicMin((unsigned int*)addr, __float_as_uint(v));
}

// ---------------- per-layer init: segment buffers + BN accumulators ----------------
__global__ void k_init(float* __restrict__ segmax, float* __restrict__ den,
                       float* __restrict__ num, float* __restrict__ sums,
                       float* __restrict__ sumsq) {
    int i = blockIdx.x * blockDim.x + threadIdx.x;
    if (i < N_NODES * C) {
        segmax[i] = -__builtin_inff();
        den[i]    = 0.0f;
        num[i]    = 0.0f;
    }
    if (i < C2) { sums[i] = 0.0f; sumsq[i] = 0.0f; }
}

// ------- weights f32 -> bf16, pre-swizzled into WMMA fragment order -------
// Fragment storage: [ks][col_tile][lane(32)][e(16)] -> each lane's 16 bf16
// elements are one contiguous, 32B-aligned chunk => 2x global_load_b128/lane.
__global__ void k_swizzle_weights(const float* __restrict__ W1, const float* __restrict__ W2,
                                  int layer, __bf16* __restrict__ W1f,
                                  __bf16* __restrict__ W2f) {
    int i = blockIdx.x * blockDim.x + threadIdx.x;   // 8192 elements each
    if (i >= C * C2) return;
    int e    = i & 15;
    int lane = (i >> 4) & 31;
    int half = lane >> 4;
    int lc   = lane & 15;
    {   // GEMM1 fragments: ks in [0,2), 8 column tiles of W1[64,128]
        int tile = (i >> 9) & 7, ks = i >> 12;
        int kk = fragK(e, half) + ks * 32;
        W1f[i] = f2bf(W1[layer * C * C2 + kk * C2 + tile * 16 + lc]);
    }
    {   // GEMM2 fragments: ks in [0,4), 4 column tiles of W2[128,64]
        int tile = (i >> 9) & 3, ks = i >> 11;
        int kk = fragK(e, half) + ks * 32;
        W2f[i] = f2bf(W2[layer * C2 * C + kk * C + tile * 16 + lc]);
    }
}

// ---------------- edge pass 1: per-(dst,channel) segment max of logits ----------------
// 64 consecutive threads handle one edge's 64 channels -> fully coalesced gather of x[src]
__global__ void k_edge_max(const float* __restrict__ x, const long long* __restrict__ ei,
                           const float* __restrict__ t, int layer,
                           float* __restrict__ segmax) {
    int tid = blockIdx.x * blockDim.x + threadIdx.x;  // E*C = 51.2M < 2^31
    if (tid >= E_EDGES * C) return;
    int e = tid >> 6, c = tid & 63;
    int src = (int)ei[e];
    int dst = (int)ei[E_EDGES + e];
    float m  = fmaxf(x[src * C + c], 0.0f) + EPS_MSG;
    float tl = t[layer];
    atomic_max_f32(&segmax[dst * C + c], m * tl);
}

// ---------------- edge pass 2: exp-sum (den) and weighted sum (num) ----------------
// x is 12.8MB -> resident in 192MB L2, so recomputing m beats spilling a 205MB tensor
__global__ void k_edge_sum(const float* __restrict__ x, const long long* __restrict__ ei,
                           const float* __restrict__ t, int layer,
                           const float* __restrict__ segmax,
                           float* __restrict__ den, float* __restrict__ num) {
    int tid = blockIdx.x * blockDim.x + threadIdx.x;
    if (tid >= E_EDGES * C) return;
    int e = tid >> 6, c = tid & 63;
    int src = (int)ei[e];
    int dst = (int)ei[E_EDGES + e];
    float m  = fmaxf(x[src * C + c], 0.0f) + EPS_MSG;
    float tl = t[layer];
    float ex = __expf(m * tl - segmax[dst * C + c]);
    atomicAdd(&den[dst * C + c], ex);
    atomicAdd(&num[dst * C + c], m * ex);
}

// ---- GEMM1: h1[N,128] = (aggr + x) @ W1 + b1, fused with combine + BN stats ----
// block = 256 thr = 8 waves; one 16x16 WMMA tile per wave covers 16x128 output.
__global__ void __launch_bounds__(256) k_gemm1(const float* __restrict__ xin,
                                               const float* __restrict__ den,
                                               const float* __restrict__ num,
                                               const __bf16* __restrict__ W1f,
                                               const float* __restrict__ b1, int layer,
                                               float* __restrict__ h1,
                                               float* __restrict__ sums,
                                               float* __restrict__ sumsq) {
    __shared__ __bf16 As[16 * C];              // 16x64 bf16 A tile (2KB)
    int m0 = blockIdx.x * 16;                  // 50000/16 = 3125 exact -> no tail
    for (int i = threadIdx.x; i < 16 * C; i += 256) {
        int gi = m0 * C + i;                   // contiguous 4KB tile reads x3
        float aggr = num[gi] / (den[gi] + DEN_EPS);
        As[i] = f2bf(aggr + xin[gi]);          // fused softmax-combine + residual
    }
    __syncthreads();

    int wave = threadIdx.x >> 5;
    int lane = threadIdx.x & 31;
    int half = lane >> 4;
    int lc   = lane & 15;
    int n0   = wave * 16;
    const v16bf* Bf = (const v16bf*)W1f;       // fragment-order weights

    v8f acc;
    float bv = b1[layer * C2 + n0 + lc];       // bias pre-loaded into f32 accumulator
    #pragma unroll
    for (int r = 0; r < 8; ++r) acc[r] = bv;

    #pragma unroll
    for (int ks = 0; ks < 2; ++ks) {           // K = 64 = 2 x 32
        v16bf a, b;
        b = Bf[(ks * 8 + wave) * 32 + lane];   // one 32B vector load per fragment
        #pragma unroll
        for (int e2 = 0; e2 < 16; ++e2)
            a[e2] = As[lc * C + fragK(e2, half) + ks * 32];  // -> ds_load_b128 x2
        acc = __builtin_amdgcn_wmma_f32_16x16x32_bf16(false, a, false, b,
                                                      (short)0, acc, false, false);
    }

    float s = 0.0f, ss = 0.0f;
    #pragma unroll
    for (int r = 0; r < 8; ++r) {              // D: M = r + 8*half, N = lane%16
        float v = acc[r];
        s += v; ss += v * v;
        h1[(m0 + r + half * 8) * C2 + n0 + lc] = v;
    }
    // lanes l and l+16 hold the same output column -> one shuffle closes the tile
    s  += __shfl_down(s, 16);
    ss += __shfl_down(ss, 16);
    if (lane < 16) {                           // fused BN partial stats
        atomicAdd(&sums[n0 + lc], s);
        atomicAdd(&sumsq[n0 + lc], ss);
    }
}

// ---------------- fold BN into per-channel scale/shift ----------------
__global__ void k_bn_finalize(const float* __restrict__ sums, const float* __restrict__ sumsq,
                              const float* __restrict__ gamma, const float* __restrict__ beta,
                              int layer, float* __restrict__ scale, float* __restrict__ shift) {
    int c = threadIdx.x;
    if (c < C2) {
        float inv_n = 1.0f / (float)N_NODES;
        float mean  = sums[c] * inv_n;
        float var   = sumsq[c] * inv_n - mean * mean;  // biased variance (torch BN)
        float sc    = gamma[layer * C2 + c] * rsqrtf(var + BN_EPS);
        scale[c] = sc;
        shift[c] = beta[layer * C2 + c] - mean * sc;
    }
}

// -------- GEMM2: out[N,64] = relu( relu(BN(h1)) @ W2 + b2 ) (bf16 WMMA) --------
// block = 128 thr = 4 waves; BN + ReLU fused into bf16 A staging.
__global__ void __launch_bounds__(128) k_gemm2(const float* __restrict__ h1,
                                               const __bf16* __restrict__ W2f,
                                               const float* __restrict__ b2,
                                               const float* __restrict__ scale,
                                               const float* __restrict__ shift, int layer,
                                               float* __restrict__ xout) {
    __shared__ __bf16 As[16 * C2];             // 16x128 bf16 A tile (4KB)
    int m0 = blockIdx.x * 16;
    for (int i = threadIdx.x; i < 16 * C2; i += 128) {
        int cc = i & (C2 - 1);
        float v = h1[m0 * C2 + i] * scale[cc] + shift[cc];  // BN
        As[i] = f2bf(fmaxf(v, 0.0f));                       // ReLU, f32->bf16
    }
    __syncthreads();

    int wave = threadIdx.x >> 5;
    int lane = threadIdx.x & 31;
    int half = lane >> 4;
    int lc   = lane & 15;
    int n0   = wave * 16;
    const v16bf* Bf = (const v16bf*)W2f;

    v8f acc;
    float bv = b2[layer * C + n0 + lc];
    #pragma unroll
    for (int r = 0; r < 8; ++r) acc[r] = bv;

    #pragma unroll
    for (int ks = 0; ks < 4; ++ks) {           // K = 128 = 4 x 32
        v16bf a, b;
        b = Bf[(ks * 4 + wave) * 32 + lane];
        #pragma unroll
        for (int e2 = 0; e2 < 16; ++e2)
            a[e2] = As[lc * C2 + fragK(e2, half) + ks * 32];
        acc = __builtin_amdgcn_wmma_f32_16x16x32_bf16(false, a, false, b,
                                                      (short)0, acc, false, false);
    }

    #pragma unroll
    for (int r = 0; r < 8; ++r) {
        int m = m0 + r + half * 8;
        xout[m * C + n0 + lc] = fmaxf(acc[r], 0.0f);        // layer-end ReLU
    }
}

// ---------------- host-side orchestration ----------------
extern "C" void kernel_launch(void* const* d_in, const int* in_sizes, int n_in,
                              void* d_out, int out_size, void* d_ws, size_t ws_size,
                              hipStream_t stream) {
    const float*     x_in  = (const float*)d_in[0];
    const long long* ei    = (const long long*)d_in[1];  // edge_index is int64 in reference
    const float*     W1    = (const float*)d_in[2];
    const float*     b1    = (const float*)d_in[3];
    const float*     gamma = (const float*)d_in[4];
    const float*     beta  = (const float*)d_in[5];
    const float*     W2    = (const float*)d_in[6];
    const float*     b2    = (const float*)d_in[7];
    const float*     t     = (const float*)d_in[8];
    float*           out   = (float*)d_out;

    // workspace carve-up (all offsets 256B aligned; total ~77MB)
    const size_t NC4 = (size_t)N_NODES * C * 4;     // 12.8 MB
    const size_t NH4 = (size_t)N_NODES * C2 * 4;    // 25.6 MB
    char* ws = (char*)d_ws;
    float*  segmax = (float*)(ws);
    float*  den    = (float*)(ws + 1 * NC4);
    float*  num    = (float*)(ws + 2 * NC4);
    float*  xbuf   = (float*)(ws + 3 * NC4);
    float*  h1     = (float*)(ws + 4 * NC4);
    char*   tail   = ws + 4 * NC4 + NH4;
    float*  sums   = (float*)(tail);
    float*  sumsq  = (float*)(tail + 512);
    float*  scale  = (float*)(tail + 1024);
    float*  shift  = (float*)(tail + 1536);
    __bf16* W1f    = (__bf16*)(tail + 2048);                  // 16KB, fragment order
    __bf16* W2f    = (__bf16*)(tail + 2048 + C * C2 * 2);     // 16KB, fragment order

    const int nc_blocks   = (N_NODES * C + 255) / 256;   // 12500
    const int edge_blocks = (E_EDGES * C + 255) / 256;   // 200000
    const int gemm_blocks = N_NODES / 16;                // 3125 (exact)

    for (int l = 0; l < N_AGGR; ++l) {
        const float* xin  = (l == 0) ? x_in : xbuf;
        float*       xout = (l == N_AGGR - 1) ? out : xbuf;

        k_init           <<<nc_blocks,   256, 0, stream>>>(segmax, den, num, sums, sumsq);
        k_swizzle_weights<<<32,          256, 0, stream>>>(W1, W2, l, W1f, W2f);
        k_edge_max       <<<edge_blocks, 256, 0, stream>>>(xin, ei, t, l, segmax);
        k_edge_sum       <<<edge_blocks, 256, 0, stream>>>(xin, ei, t, l, segmax, den, num);
        k_gemm1          <<<gemm_blocks, 256, 0, stream>>>(xin, den, num, W1f, b1, l,
                                                           h1, sums, sumsq);
        k_bn_finalize    <<<1,           128, 0, stream>>>(sums, sumsq, gamma, beta, l,
                                                           scale, shift);
        k_gemm2          <<<gemm_blocks, 128, 0, stream>>>(h1, W2f, b2, scale, shift, l, xout);
    }
}
